// TemporalBlock_3066606649424
// MI455X (gfx1250) — compile-verified, matrix operand order checked by reading, not run on hardware
//
#include <hip/hip_runtime.h>
#include <hip/hip_bf16.h>
#include <math.h>

#define DEVINL static __device__ __forceinline__

typedef __attribute__((ext_vector_type(16))) __bf16 v16bf;
typedef __attribute__((ext_vector_type(8)))  __bf16 v8bf;
typedef __attribute__((ext_vector_type(8)))  float  v8f;

// ---------- helpers ----------

DEVINL __bf16 f2bf(float f) {
  unsigned u = __builtin_bit_cast(unsigned, f);
  unsigned r = u + 0x7FFFu + ((u >> 16) & 1u);   // round-to-nearest-even
  return __builtin_bit_cast(__bf16, (unsigned short)(r >> 16));
}

// A fragment: 16x32 bf16 row-major (ld elems/row). Lane layout (ISA 7.12.2):
// lanes 0-15 hold K 0..7 & 16..23, lanes 16-31 hold K 8..15 & 24..31
// -> exactly two contiguous 16B runs per lane.
DEVINL v16bf load_a_frag(const __bf16* __restrict__ base, int ld, int lane) {
  const int m = lane & 15;
  const int kb = (lane >> 4) << 3;           // 0 or 8
  const __bf16* p = base + m * ld + kb;
  v8bf lo = *(const v8bf*)(p);               // K = kb .. kb+7
  v8bf hi = *(const v8bf*)(p + 16);          // K = kb+16 .. kb+23
  return __builtin_shufflevector(lo, hi, 0, 1, 2, 3, 4, 5, 6, 7,
                                 8, 9, 10, 11, 12, 13, 14, 15);
}

// B fragment from fragment-major tile: [lane][16 elems] contiguous (1KB/tile)
DEVINL v16bf load_frag(const __bf16* __restrict__ tile, int lane) {
  return *(const v16bf*)(tile + (lane << 4));
}

// fragment-major position of logical (k in 0..31, n in 0..15) inside a tile
DEVINL int frag_pos(int k, int n) {
  int lane = n + (((k >> 3) & 1) << 4);
  int e = ((k >> 4) << 3) + (k & 7);
  return (lane << 4) + e;
}

DEVINL v8f wmma_bf16(v16bf a, v16bf b, v8f c) {
  return __builtin_amdgcn_wmma_f32_16x16x32_bf16(false, a, false, b, (short)0, c,
                                                 false, false);
}

// ---------- CDNA5 async DMA: global -> LDS, 16B per lane ----------

DEVINL void async_ld_b128(void* smem, const void* gmem) {
  unsigned lds = (unsigned)(size_t)smem;   // LDS aperture: addr[31:0] = offset
  asm volatile("global_load_async_to_lds_b128 %0, %1, off"
               :: "v"(lds), "v"(gmem) : "memory");
}
DEVINL void async_wait() {
  asm volatile("s_wait_asynccnt 0x0" ::: "memory");
}

// ---------- kernel 0: fp32 -> bf16 cast + swizzle into fragment-major ----------
// out layout: [ks = K/32][nt = N/16][lane][e] ; in: row-major [K][N]

__global__ __launch_bounds__(256) void cast_swizzle_kernel(
    const float* __restrict__ in, __bf16* __restrict__ out, int K, int N) {
  int i = blockIdx.x * 256 + threadIdx.x;
  if (i >= K * N) return;
  int e = i & 15, lane = (i >> 4) & 31, tile = i >> 9;
  int ntiles = N >> 4;
  int ks = tile / ntiles, nt = tile % ntiles;
  int k = ks * 32 + ((e >> 3) << 4) + ((lane >> 4) << 3) + (e & 7);
  int n = nt * 16 + (lane & 15);
  out[i] = f2bf(in[(size_t)k * N + n]);
}

// ---------- kernel 1: qkv = x @ qkv_w + b ; q,k,v all stored [b][t][64] ----------

__global__ __launch_bounds__(256) void qkv_kernel(
    const float* __restrict__ x, const __bf16* __restrict__ wq,
    const float* __restrict__ bias,
    __bf16* __restrict__ q, __bf16* __restrict__ k, __bf16* __restrict__ v) {
  __shared__ __attribute__((aligned(32))) __bf16 sX[32 * 96];
  __shared__ __attribute__((aligned(32))) __bf16 sW[3 * 12 * 512];  // frag-major
  const int tid = threadIdx.x, lane = tid & 31, wid = tid >> 5;
  const int row0 = blockIdx.x * 32;        // 2048 % 32 == 0 -> no batch straddle
  const int bb = row0 >> 11, tbase = row0 & 2047;   // uniform per block
  // async DMA of the (pre-swizzled) weight panel
  for (int i = tid * 8; i < 3 * 12 * 512; i += 256 * 8) async_ld_b128(sW + i, wq + i);
  for (int i = tid; i < 32 * 96; i += 256) sX[i] = f2bf(x[(size_t)row0 * 96 + i]);
  async_wait();
  __syncthreads();
  const int colc = lane & 15, rbase = (lane >> 4) << 3;
  #pragma unroll
  for (int ti = 0; ti < 3; ++ti) {   // 24 tiles (2 Mt x 12 Nt) over 8 waves
    int tile = wid * 3 + ti;
    int mt = tile / 12, nt = tile % 12;
    v8f acc = {};
    #pragma unroll
    for (int ks = 0; ks < 3; ++ks) {
      v16bf a = load_a_frag(sX + (mt * 16) * 96 + ks * 32, 96, lane);
      v16bf b = load_frag(sW + (ks * 12 + nt) * 512, lane);
      acc = wmma_bf16(a, b, acc);
    }
    // tile-uniform destination select (scalar branch via readfirstlane)
    int ntu = __builtin_amdgcn_readfirstlane(nt);
    __bf16* dst = (ntu < 4) ? q : (ntu < 8) ? k : v;
    int nloc = nt * 16 + colc - ((ntu < 4) ? 0 : (ntu < 8) ? 64 : 128);
    float bn = bias[nt * 16 + colc];
    __bf16* base = dst + ((size_t)bb * 2048 + tbase + mt * 16 + rbase) * 64 + nloc;
    #pragma unroll
    for (int i = 0; i < 8; ++i) base[(size_t)i * 64] = f2bf(acc[i] + bn);
  }
}

// ---------- kernel 2: flash attention, 64 query rows/block, key step 32 ----------

__global__ __launch_bounds__(256) void attn_kernel(
    const __bf16* __restrict__ q, const __bf16* __restrict__ k,
    const __bf16* __restrict__ v, const float* __restrict__ scale,
    __bf16* __restrict__ attn) {
  __shared__ __attribute__((aligned(32))) __bf16 sQ[64 * 64];
  __shared__ __attribute__((aligned(32))) __bf16 sKf[4 * 512];  // [ks2][nh2] frag
  __shared__ __attribute__((aligned(32))) __bf16 sVf[4 * 512];  // [nt4] frag
  __shared__ __attribute__((aligned(32))) __bf16 sP[4 * 16 * 32];
  __shared__ float sS[4 * 16 * 32];
  __shared__ float sM[64], sL[64], sAl[64];
  const int tid = threadIdx.x, lane = tid & 31, wid = tid >> 5;
  const int b = blockIdx.y, t0 = blockIdx.x * 64;
  const float scl = scale[0];
  const __bf16* kb = k + (size_t)b * 2048 * 64;
  const __bf16* vb = v + (size_t)b * 2048 * 64;
  // Q tile: contiguous 8KB, async DMA
  {
    const __bf16* gq = q + ((size_t)b * 2048 + t0) * 64;
    for (int i = tid * 8; i < 64 * 64; i += 256 * 8) async_ld_b128(sQ + i, gq + i);
  }
  if (tid < 64) { sM[tid] = -3.0e38f; sL[tid] = 0.f; }
  async_wait();
  const int mt = wid >> 1, nh = wid & 1;      // wave owns (Mt, key/output half)
  const int colc = lane & 15, rbase = (lane >> 4) << 3;
  v8f o0 = {}, o1 = {};
  for (int s0 = 0; s0 < 2048; s0 += 32) {
    __syncthreads();  // protect previous iteration's consumers
    // K tile: coalesced read of k[b][s0+j][d], transpose absorbed by frag scatter
    for (int i = tid; i < 32 * 64; i += 256) {
      int j = i >> 6, d = i & 63;
      __bf16 val = kb[(size_t)(s0 + j) * 64 + d];
      sKf[((((d >> 5) << 1) | (j >> 4)) << 9) + frag_pos(d & 31, j & 15)] = val;
    }
    // V tile: coalesced read, scatter-store into fragment-major [nt]
    for (int i = tid; i < 32 * 64; i += 256) {
      int j = i >> 6, d = i & 63;
      __bf16 val = vb[(size_t)(s0 + j) * 64 + d];
      sVf[((d >> 4) << 9) + frag_pos(j, d & 15)] = val;
    }
    __syncthreads();
    // S = Q K^T for this wave's 16x16 tile (2 WMMA over D=64)
    v8f sc = {};
    {
      v16bf a0 = load_a_frag(sQ + (mt * 16) * 64 + 0, 64, lane);
      v16bf b0 = load_frag(sKf + (0 * 2 + nh) * 512, lane);
      sc = wmma_bf16(a0, b0, sc);
      v16bf a1 = load_a_frag(sQ + (mt * 16) * 64 + 32, 64, lane);
      v16bf b1 = load_frag(sKf + (1 * 2 + nh) * 512, lane);
      sc = wmma_bf16(a1, b1, sc);
    }
    #pragma unroll
    for (int i = 0; i < 8; ++i)
      sS[(mt * 16 + rbase + i) * 32 + nh * 16 + colc] = sc[i] * scl;
    __syncthreads();
    // online-softmax stats + P (one thread per query row)
    if (tid < 64) {
      const float* srow = &sS[tid * 32];
      float mx = sM[tid];
      #pragma unroll
      for (int j = 0; j < 32; ++j) mx = fmaxf(mx, srow[j]);
      float al = __expf(sM[tid] - mx);
      float sum = 0.f;
      __bf16* prow = &sP[tid * 32];
      #pragma unroll
      for (int j = 0; j < 32; ++j) {
        float e = __expf(srow[j] - mx);
        sum += e;
        prow[j] = f2bf(e);
      }
      sL[tid] = sL[tid] * al + sum;
      sM[tid] = mx;
      sAl[tid] = al;
    }
    __syncthreads();
    // O = O*alpha + P @ V (wave's two 16-col output tiles)
    v16bf pa  = load_a_frag(sP + (mt * 16) * 32, 32, lane);
    v16bf vb0 = load_frag(sVf + (nh * 2 + 0) * 512, lane);
    v16bf vb1 = load_frag(sVf + (nh * 2 + 1) * 512, lane);
    #pragma unroll
    for (int i = 0; i < 8; ++i) {
      float al = sAl[mt * 16 + rbase + i];
      o0[i] *= al;
      o1[i] *= al;
    }
    o0 = wmma_bf16(pa, vb0, o0);
    o1 = wmma_bf16(pa, vb1, o1);
  }
  __syncthreads();
  #pragma unroll
  for (int i = 0; i < 8; ++i) {
    int r = mt * 16 + rbase + i;
    float inv = 1.f / sL[r];
    int t = t0 + r;
    attn[((size_t)b * 2048 + t) * 64 + (nh * 2 + 0) * 16 + colc] = f2bf(o0[i] * inv);
    attn[((size_t)b * 2048 + t) * 64 + (nh * 2 + 1) * 16 + colc] = f2bf(o1[i] * inv);
  }
}

// ---------- kernel 3: conv1 (K=3, Cin=64 -> Cout=128) + bias + ReLU ----------

__global__ __launch_bounds__(256) void conv1_kernel(
    const __bf16* __restrict__ attn, const __bf16* __restrict__ w1,
    const float* __restrict__ b1, __bf16* __restrict__ h) {
  __shared__ __attribute__((aligned(32))) __bf16 sA[66 * 64];     // halo tile
  __shared__ __attribute__((aligned(32))) __bf16 sW[6 * 8 * 512]; // frag-major
  const int tid = threadIdx.x, lane = tid & 31, wid = tid >> 5;
  const int b = blockIdx.y, t0 = blockIdx.x * 64;
  // interior 64 rows are contiguous in global: async DMA into rows 1..64
  {
    const __bf16* g = attn + ((size_t)b * 2048 + t0) * 64;
    for (int i = tid * 8; i < 64 * 64; i += 256 * 8) async_ld_b128(sA + 64 + i, g + i);
  }
  for (int i = tid * 8; i < 6 * 8 * 512; i += 256 * 8) async_ld_b128(sW + i, w1 + i);
  // halo rows 0 (t0-1) and 65 (t0+64), zero-padded at sequence edges
  for (int i = tid; i < 2 * 64; i += 256) {
    int r = (i >> 6) ? 65 : 0, c = i & 63, t = t0 + r - 1;
    sA[r * 64 + c] = (t >= 0 && t < 2048) ? attn[((size_t)b * 2048 + t) * 64 + c]
                                          : f2bf(0.f);
  }
  async_wait();
  __syncthreads();
  const int colc = lane & 15, rbase = (lane >> 4) << 3;
  v8f acc[4] = {};
  for (int ks = 0; ks < 6; ++ks) {     // K = 192; step never straddles a tap
    int tap = ks >> 1, cb = (ks & 1) * 32;
    v16bf bf = load_frag(sW + (ks * 8 + wid) * 512, lane);
    #pragma unroll
    for (int mt = 0; mt < 4; ++mt) {
      v16bf af = load_a_frag(sA + (mt * 16 + tap) * 64 + cb, 64, lane);
      acc[mt] = wmma_bf16(af, bf, acc[mt]);
    }
  }
  int n = wid * 16 + colc;
  float bn = b1[n];
  __bf16* base = h + ((size_t)b * 2048 + t0 + rbase) * 128 + n;
  #pragma unroll
  for (int mt = 0; mt < 4; ++mt) {
    #pragma unroll
    for (int i = 0; i < 8; ++i) {
      float val = fmaxf(acc[mt][i] + bn, 0.f);
      base[(size_t)(mt * 16 + i) * 128] = f2bf(val);
    }
  }
}

// ---------- kernel 4: resid = (1 + 1/T) * (x @ down_w) ----------
// sum_s softmax_s(scores) == 1 exactly -> softmax over T of ones == 1/T,
// so the "enhanced residual" collapses into the skip path.

__global__ __launch_bounds__(256) void resid_kernel(
    const float* __restrict__ x, const __bf16* __restrict__ wd,
    float* __restrict__ resid) {
  __shared__ __attribute__((aligned(32))) __bf16 sX[64 * 96];
  __shared__ __attribute__((aligned(32))) __bf16 sW[3 * 8 * 512]; // frag-major
  const int tid = threadIdx.x, lane = tid & 31, wid = tid >> 5;
  const int row0 = blockIdx.x * 64;
  for (int i = tid * 8; i < 3 * 8 * 512; i += 256 * 8) async_ld_b128(sW + i, wd + i);
  for (int i = tid; i < 64 * 96; i += 256) sX[i] = f2bf(x[(size_t)row0 * 96 + i]);
  async_wait();
  __syncthreads();
  const int colc = lane & 15, rbase = (lane >> 4) << 3;
  const float fac = 1.0f + 1.0f / 2048.0f;
  v8f acc[4] = {};
  for (int ks = 0; ks < 3; ++ks) {
    v16bf bf = load_frag(sW + (ks * 8 + wid) * 512, lane);
    #pragma unroll
    for (int mt = 0; mt < 4; ++mt) {
      v16bf af = load_a_frag(sX + (mt * 16) * 96 + ks * 32, 96, lane);
      acc[mt] = wmma_bf16(af, bf, acc[mt]);
    }
  }
  int n = wid * 16 + colc;
  float* base = resid + ((size_t)row0 + rbase) * 128 + n;
  #pragma unroll
  for (int mt = 0; mt < 4; ++mt) {
    #pragma unroll
    for (int i = 0; i < 8; ++i)
      base[(size_t)(mt * 16 + i) * 128] = acc[mt][i] * fac;
  }
}

// ---------- kernel 5: conv2 + bias + ReLU, then + resid, final ReLU ----------

__global__ __launch_bounds__(256) void conv2_kernel(
    const __bf16* __restrict__ h, const __bf16* __restrict__ w2,
    const float* __restrict__ b2, const float* __restrict__ resid,
    float* __restrict__ out) {
  __shared__ __attribute__((aligned(32))) __bf16 sA[66 * 128];
  __shared__ __attribute__((aligned(32))) __bf16 sW[12 * 2 * 512]; // 32-col slice
  const int tid = threadIdx.x, lane = tid & 31, wid = tid >> 5;
  const int b = blockIdx.y, t0 = blockIdx.x * 64, z = blockIdx.z; // n0 = z*32
  // interior rows async
  {
    const __bf16* g = h + ((size_t)b * 2048 + t0) * 128;
    for (int i = tid * 8; i < 64 * 128; i += 256 * 8) async_ld_b128(sA + 128 + i, g + i);
  }
  // weight slice: for each ks, tiles 2z and 2z+1 are 1024 contiguous elems
  for (int ks = 0; ks < 12; ++ks) {
    const __bf16* g = w2 + (size_t)(ks * 8 + 2 * z) * 512;
    __bf16* s = sW + ks * 1024;
    for (int i = tid * 8; i < 1024; i += 256 * 8) async_ld_b128(s + i, g + i);
  }
  for (int i = tid; i < 2 * 128; i += 256) {
    int r = (i >> 7) ? 65 : 0, c = i & 127, t = t0 + r - 1;
    sA[r * 128 + c] = (t >= 0 && t < 2048) ? h[((size_t)b * 2048 + t) * 128 + c]
                                           : f2bf(0.f);
  }
  async_wait();
  __syncthreads();
  const int colc = lane & 15, rbase = (lane >> 4) << 3;
  const int mt = wid >> 1, nt = wid & 1;     // 4 Mt x 2 Nt tiles over 8 waves
  v8f acc = {};
  for (int ks = 0; ks < 12; ++ks) {          // K = 384
    int tap = ks >> 2, cb = (ks & 3) * 32;
    v16bf af = load_a_frag(sA + (mt * 16 + tap) * 128 + cb, 128, lane);
    v16bf bf = load_frag(sW + (ks * 2 + nt) * 512, lane);
    acc = wmma_bf16(af, bf, acc);
  }
  int n = z * 32 + nt * 16 + colc;
  float bn = b2[n];
  const size_t base = ((size_t)b * 2048 + t0 + mt * 16 + rbase) * 128 + n;
  #pragma unroll
  for (int i = 0; i < 8; ++i) {
    size_t idx = base + (size_t)i * 128;
    float conv = fmaxf(acc[i] + bn, 0.f);
    out[idx] = fmaxf(conv + resid[idx], 0.f);
  }
}

// ---------- launcher ----------

extern "C" void kernel_launch(void* const* d_in, const int* in_sizes, int n_in,
                              void* d_out, int out_size, void* d_ws, size_t ws_size,
                              hipStream_t stream) {
  (void)in_sizes; (void)n_in; (void)out_size; (void)ws_size;
  const float* x       = (const float*)d_in[0];  // [16,2048,96]
  const float* qkv_w   = (const float*)d_in[1];  // [96,192]
  const float* qkv_b   = (const float*)d_in[2];  // [192]
  const float* scale   = (const float*)d_in[3];  // [1]
  const float* conv_w1 = (const float*)d_in[4];  // [3,64,128] -> [192,128]
  const float* conv_b1 = (const float*)d_in[5];  // [128]
  const float* conv_w2 = (const float*)d_in[6];  // [3,128,128] -> [384,128]
  const float* conv_b2 = (const float*)d_in[7];  // [128]
  const float* down_w  = (const float*)d_in[8];  // [96,128]
  float* out = (float*)d_out;

  const int B = 16, T = 2048, D = 64, NO = 128;
  size_t off = 0;
  auto alloc = [&](size_t bytes) {
    void* p = (char*)d_ws + off;
    off += (bytes + 255) & ~(size_t)255;
    return p;
  };
  __bf16* qb     = (__bf16*)alloc((size_t)B * T * D * 2);
  __bf16* kbuf   = (__bf16*)alloc((size_t)B * T * D * 2);
  __bf16* vbuf   = (__bf16*)alloc((size_t)B * T * D * 2);
  __bf16* attnb  = (__bf16*)alloc((size_t)B * T * D * 2);
  __bf16* hb     = (__bf16*)alloc((size_t)B * T * NO * 2);
  float*  residf = (float*)alloc((size_t)B * T * NO * 4);
  __bf16* wqb    = (__bf16*)alloc((size_t)96 * 192 * 2);
  __bf16* w1b    = (__bf16*)alloc((size_t)192 * 128 * 2);
  __bf16* w2b    = (__bf16*)alloc((size_t)384 * 128 * 2);
  __bf16* wdb    = (__bf16*)alloc((size_t)96 * 128 * 2);

  // weight casts + swizzle into WMMA fragment-major layout
  cast_swizzle_kernel<<<(96 * 192 + 255) / 256, 256, 0, stream>>>(qkv_w, wqb, 96, 192);
  cast_swizzle_kernel<<<(192 * 128 + 255) / 256, 256, 0, stream>>>(conv_w1, w1b, 192, 128);
  cast_swizzle_kernel<<<(384 * 128 + 255) / 256, 256, 0, stream>>>(conv_w2, w2b, 384, 128);
  cast_swizzle_kernel<<<(96 * 128 + 255) / 256, 256, 0, stream>>>(down_w, wdb, 96, 128);

  // qkv projection + split (q,k,v all [b][t][64])
  qkv_kernel<<<(B * T) / 32, 256, 0, stream>>>(x, wqb, qkv_b, qb, kbuf, vbuf);

  // flash attention
  attn_kernel<<<dim3(T / 64, B), 256, 0, stream>>>(qb, kbuf, vbuf, scale, attnb);

  // conv1 + ReLU
  conv1_kernel<<<dim3(T / 64, B), 256, 0, stream>>>(attnb, w1b, conv_b1, hb);

  // residual path (skip + enhanced residual, collapsed analytically)
  resid_kernel<<<(B * T) / 64, 256, 0, stream>>>(x, wdb, residf);

  // conv2 + bias + ReLU + residual + final ReLU
  conv2_kernel<<<dim3(T / 64, B, NO / 32), 256, 0, stream>>>(hb, w2b, conv_b2,
                                                             residf, out);
}
